// single_scale_gnn_4209067950734
// MI455X (gfx1250) — matrix operand order repaired and checked by reading, not compiled, not run on hardware
//
#include <hip/hip_runtime.h>
#include <stdint.h>
#include <stddef.h>

typedef __bf16 bf16;
typedef __attribute__((ext_vector_type(16))) __bf16 v16bf;
typedef __attribute__((ext_vector_type(8)))  __bf16 v8bf;
typedef __attribute__((ext_vector_type(8)))  float  v8f;

__device__ __forceinline__ float gelu_f(float x) {
  return 0.5f * x * (1.0f + erff(x * 0.7071067811865475f));
}

// ---------------------------------------------------------------- elementwise
__global__ void k_f32_to_bf16(const float* __restrict__ in, bf16* __restrict__ out, int n) {
  int i = blockIdx.x * blockDim.x + threadIdx.x;
  if (i < n) out[i] = (bf16)in[i];
}

// x[B,S,C] -> mean over p strided blocks -> xp[B,L,C] (f32) and bigvar slot0 (bf16, row stride 4C)
__global__ void k_patch_mean(const float* __restrict__ x, float* __restrict__ xp,
                             bf16* __restrict__ bigvar, int B, int S, int C, int L, int p) {
  int i = blockIdx.x * blockDim.x + threadIdx.x;
  int total = B * L * C;
  if (i >= total) return;
  int c = i % C;
  int l = (i / C) % L;
  int b = i / (C * L);
  float s = 0.f;
  for (int q = 0; q < p; ++q)
    s += x[((size_t)b * S + (size_t)q * L + l) * (size_t)C + c];
  float m = s / (float)p;
  xp[i] = m;
  bigvar[((size_t)b * L + l) * (size_t)(4 * C) + c] = (bf16)m;
}

// im2col for circular conv k=3: out[(b*L+l), k*C+c] = xp[b, (l+k-1) mod L, c]  (bf16)
__global__ void k_gather_conv_rows(const float* __restrict__ xp, bf16* __restrict__ out,
                                   int B, int L, int C) {
  int i = blockIdx.x * blockDim.x + threadIdx.x;
  int total = B * L * 3 * C;
  if (i >= total) return;
  int within = i % (3 * C);
  int row = i / (3 * C);
  int k = within / C;
  int c = within % C;
  int l = row % L;
  int b = row / L;
  int l2 = l + k - 1; if (l2 < 0) l2 += L; if (l2 >= L) l2 -= L;
  out[i] = (bf16)xp[((size_t)b * L + l2) * (size_t)C + c];
}

// -------------------------------------------------------------------- WMMA GEMM
// C[m,n] = sum_k A[m,k]*B[k,n].  128 threads = 4 waves; block tile 64x64, each
// wave owns a 32x32 quadrant = 2x2 WMMA (bf16 16x16x32, f32 accum).
// A tile staged row-major in LDS, B tile staged TRANSPOSED ([n][k]) so both
// fragment types are contiguous b128 LDS reads.
// ep: 0=none, 1=gelu, 2=+pos_embed(Lpe,Dpe), 3=+bias then gelu.
// Writes f32 (Cf) and/or bf16 (Cb). M,N multiples of 64; K multiple of 32.
__global__ void __launch_bounds__(128)
k_gemm_bf16(const bf16* __restrict__ A, long long lda, long long bsA,
            const bf16* __restrict__ Bm, long long ldb, long long bsB,
            float* __restrict__ Cf, long long ldcf, long long bsCf,
            bf16* __restrict__ Cb, long long ldcb, long long bsCb,
            int M, int N, int K,
            const float* __restrict__ bias, int ep, int Lpe, int Dpe) {
  __shared__ bf16 Alds[64 * 32];   // [m][k] row-major
  __shared__ bf16 Blds[64 * 32];   // [n][k] (transposed)

  int tid  = threadIdx.x;
  int wid  = tid >> 5;
  int lane = tid & 31;
  int wy = wid >> 1, wx = wid & 1;          // wave quadrant
  int m0 = (int)blockIdx.y * 64;
  int n0 = (int)blockIdx.x * 64;
  int bz = blockIdx.z;
  const bf16* Ab = A + (size_t)bz * (size_t)bsA;
  const bf16* Bb = Bm + (size_t)bz * (size_t)bsB;

  int half = lane >> 4;                     // 0/1
  int sub  = lane & 15;

  // staging indices
  int am = tid >> 1;                        // 0..63 : A-tile row
  int ak = (tid & 1) * 16;                  // 0/16  : A-tile k offset
  int bk = tid >> 2;                        // 0..31 : B-tile k row
  int bn = (tid & 3) * 16;                  // 0/16/32/48 : B-tile col group

  v8f acc[2][2] = {{{}, {}}, {{}, {}}};

  for (int k0 = 0; k0 < K; k0 += 32) {
    // --- fetch next tiles from global (b128 vector loads) ---
    const bf16* ag = Ab + (size_t)(m0 + am) * (size_t)lda + (k0 + ak);
    v8bf a0 = *(const v8bf*)ag;
    v8bf a1 = *(const v8bf*)(ag + 8);
    const bf16* bg = Bb + (size_t)(k0 + bk) * (size_t)ldb + (n0 + bn);
    v8bf b0 = *(const v8bf*)bg;
    v8bf b1 = *(const v8bf*)(bg + 8);

    __syncthreads();                        // previous iter's LDS reads done
    *(v8bf*)&Alds[am * 32 + ak]     = a0;
    *(v8bf*)&Alds[am * 32 + ak + 8] = a1;
#pragma unroll
    for (int e = 0; e < 8; ++e) {           // transpose-scatter B
      Blds[(bn + e) * 32 + bk]     = b0[e];
      Blds[(bn + 8 + e) * 32 + bk] = b1[e];
    }
    __syncthreads();

    if (k0 + 32 < K) {                      // gfx1250 global_prefetch_b8
      __builtin_prefetch((const void*)(ag + 32), 0, 1);
      __builtin_prefetch((const void*)(bg + (size_t)32 * (size_t)ldb), 0, 1);
    }

    // --- build fragments from LDS (contiguous b128 reads) ---
    v16bf afr[2], bfr[2];
#pragma unroll
    for (int ms = 0; ms < 2; ++ms) {
      int m = wy * 32 + ms * 16 + sub;
      v8bf lo = *(const v8bf*)&Alds[m * 32 + half * 8];
      v8bf hi = *(const v8bf*)&Alds[m * 32 + 16 + half * 8];
#pragma unroll
      for (int e = 0; e < 8; ++e) { afr[ms][e] = lo[e]; afr[ms][e + 8] = hi[e]; }
    }
#pragma unroll
    for (int ns = 0; ns < 2; ++ns) {
      int n = wx * 32 + ns * 16 + sub;
      v8bf lo = *(const v8bf*)&Blds[n * 32 + half * 8];
      v8bf hi = *(const v8bf*)&Blds[n * 32 + 16 + half * 8];
#pragma unroll
      for (int e = 0; e < 8; ++e) { bfr[ns][e] = lo[e]; bfr[ns][e + 8] = hi[e]; }
    }

#pragma unroll
    for (int ms = 0; ms < 2; ++ms)
#pragma unroll
      for (int ns = 0; ns < 2; ++ns)
        acc[ms][ns] = __builtin_amdgcn_wmma_f32_16x16x32_bf16(
            false, afr[ms], false, bfr[ns], (short)0, acc[ms][ns], false, false);
  }

  // --- epilogue + store ---
#pragma unroll
  for (int ms = 0; ms < 2; ++ms) {
#pragma unroll
    for (int ns = 0; ns < 2; ++ns) {
      int n = n0 + wx * 32 + ns * 16 + sub;
#pragma unroll
      for (int r = 0; r < 8; ++r) {
        int m = m0 + wy * 32 + ms * 16 + r + 8 * half;
        float v = acc[ms][ns][r];
        if (ep == 3) { if (bias) v += bias[n]; v = gelu_f(v); }
        else if (ep == 1) { v = gelu_f(v); }
        else if (ep == 2) {
          int l = m % Lpe;
          int j = n >> 1;
          float dv  = __expf(-9.2103403719761836f * (float)(2 * j) / (float)Dpe);
          float ang = (float)l * dv;
          v += (n & 1) ? __cosf(ang) : __sinf(ang);
        }
        if (Cf) Cf[(size_t)bz * (size_t)bsCf + (size_t)m * (size_t)ldcf + n] = v;
        if (Cb) Cb[(size_t)bz * (size_t)bsCb + (size_t)m * (size_t)ldcb + n] = (bf16)v;
      }
    }
  }
}

// row-softmax of G[L,L] (row t), write transposed bf16: outT[l,t]
__global__ void k_softmax_transpose(const float* __restrict__ G, bf16* __restrict__ outT, int L) {
  int t = blockIdx.x;
  int tid = threadIdx.x;
  const float* row = G + (size_t)t * L;
  __shared__ float red[256];
  float mx = -3.0e38f;
  for (int l = tid; l < L; l += 256) mx = fmaxf(mx, row[l]);
  red[tid] = mx; __syncthreads();
  for (int s = 128; s > 0; s >>= 1) { if (tid < s) red[tid] = fmaxf(red[tid], red[tid + s]); __syncthreads(); }
  mx = red[0]; __syncthreads();
  float sum = 0.f;
  for (int l = tid; l < L; l += 256) sum += __expf(row[l] - mx);
  red[tid] = sum; __syncthreads();
  for (int s = 128; s > 0; s >>= 1) { if (tid < s) red[tid] += red[tid + s]; __syncthreads(); }
  float inv = 1.0f / red[0];
  for (int l = tid; l < L; l += 256)
    outT[(size_t)l * L + t] = (bf16)(__expf(row[l] - mx) * inv);
}

// var adjacency: softmax(gelu(V@V)) -> top-6 mask (zeros elsewhere) -> softmax again -> bf16
__global__ void __launch_bounds__(64)
k_var_adj(const float* __restrict__ V, bf16* __restrict__ outA) {
  const int C = 64;
  __shared__ float Vs[64 * 64];
  __shared__ float Adj[64 * 64];
  int t = threadIdx.x;
  for (int idx = t; idx < C * C; idx += 64) Vs[idx] = V[idx];
  __syncthreads();
  float mx = -3.0e38f;
  for (int v = 0; v < C; ++v) {
    float s = 0.f;
    for (int c = 0; c < C; ++c) s += Vs[t * C + c] * Vs[c * C + v];
    s = gelu_f(s);
    Adj[t * C + v] = s;
    mx = fmaxf(mx, s);
  }
  float sum = 0.f;
  for (int v = 0; v < C; ++v) { float e = __expf(Adj[t * C + v] - mx); Adj[t * C + v] = e; sum += e; }
  float inv = 1.0f / sum;
  for (int v = 0; v < C; ++v) Adj[t * C + v] *= inv;
  unsigned long long sel = 0ull;
  for (int it = 0; it < 6; ++it) {
    float best = -1.0f; int bi = 0;
    for (int v = 0; v < C; ++v)
      if (((sel >> v) & 1ull) == 0ull && Adj[t * C + v] > best) { best = Adj[t * C + v]; bi = v; }
    sel |= (1ull << bi);
  }
  float mx2 = 0.0f;  // masked-out entries are exactly 0.0 (not -inf)
  for (int v = 0; v < C; ++v) { float val = ((sel >> v) & 1ull) ? Adj[t * C + v] : 0.f; mx2 = fmaxf(mx2, val); }
  float s2 = 0.f;
  for (int v = 0; v < C; ++v) {
    float val = ((sel >> v) & 1ull) ? Adj[t * C + v] : 0.f;
    float e = __expf(val - mx2);
    Adj[t * C + v] = e; s2 += e;
  }
  float inv2 = 1.0f / s2;
  for (int v = 0; v < C; ++v) outA[t * C + v] = (bf16)(Adj[t * C + v] * inv2);
}

// softmax over last axis of xt[B*L, S], replicated p times along the row axis:
// out[b, q*L+l, s] = softmax(xt[b*L+l, :])[s] for q in 0..p-1
__global__ void k_softmax_tile(const float* __restrict__ xt, float* __restrict__ out,
                               int L, int Sdim, int p) {
  int row = blockIdx.x;            // b*L + l
  int b = row / L, l = row % L;
  int tid = threadIdx.x;
  const float* src = xt + (size_t)row * Sdim;
  __shared__ float red[256];
  float mx = -3.0e38f;
  for (int s = tid; s < Sdim; s += 256) mx = fmaxf(mx, src[s]);
  red[tid] = mx; __syncthreads();
  for (int st = 128; st > 0; st >>= 1) { if (tid < st) red[tid] = fmaxf(red[tid], red[tid + st]); __syncthreads(); }
  mx = red[0]; __syncthreads();
  float sum = 0.f;
  for (int s = tid; s < Sdim; s += 256) sum += __expf(src[s] - mx);
  red[tid] = sum; __syncthreads();
  for (int st = 128; st > 0; st >>= 1) { if (tid < st) red[tid] += red[tid + st]; __syncthreads(); }
  float inv = 1.0f / red[0];
  for (int s = tid; s < Sdim; s += 256) {
    float val = __expf(src[s] - mx) * inv;
    for (int q = 0; q < p; ++q)
      out[((size_t)b * Sdim + (size_t)q * L + l) * (size_t)Sdim + s] = val;
  }
}

// ---------------------------------------------------------------------- driver
extern "C" void kernel_launch(void* const* d_in, const int* in_sizes, int n_in,
                              void* d_out, int out_size, void* d_ws, size_t ws_size,
                              hipStream_t stream) {
  (void)in_sizes; (void)n_in; (void)out_size; (void)ws_size;
  const float* x       = (const float*)d_in[0];
  const float* conv_w  = (const float*)d_in[1];
  const float* tvec[2] = {(const float*)d_in[2], (const float*)d_in[3]};
  const float* vvec[2] = {(const float*)d_in[4], (const float*)d_in[5]};
  const float* shv[2]  = {(const float*)d_in[6], (const float*)d_in[7]};
  const float* tg_w    = (const float*)d_in[8];
  const float* tg_b    = (const float*)d_in[9];
  const float* vg_w    = (const float*)d_in[10];
  const float* vg_b    = (const float*)d_in[11];
  float* out = (float*)d_out;

  const int B = 16, S = 1024, C = 64, D = 256;

  // ---- workspace layout (256B aligned), sized for the larger scale (L=512) ----
  char* w = (char*)d_ws;
  auto alloc = [&](size_t bytes) -> void* {
    void* r = (void*)w; w += (bytes + 255) & ~(size_t)255; return r;
  };
  float* xp_f32   = (float*)alloc((size_t)B * 512 * C * 4);          // x_patch
  bf16*  aexp_bf  = (bf16*) alloc((size_t)B * 512 * 3 * C * 2);      // im2col rows
  bf16*  convw_bf = (bf16*) alloc((size_t)3 * C * D * 2);
  bf16*  tgw_bf   = (bf16*) alloc((size_t)4 * D * C * 2);
  bf16*  vgw_bf   = (bf16*) alloc((size_t)4 * C * C * 2);
  bf16*  sh_bf    = (bf16*) alloc((size_t)C * S * 2);
  bf16*  tv_bf    = (bf16*) alloc((size_t)512 * 512 * 2);
  float* tg_f32   = (float*)alloc((size_t)512 * 512 * 4);            // gelu(T@T)
  bf16*  tadjT_bf = (bf16*) alloc((size_t)512 * 512 * 2);            // transposed adj
  bf16*  vadj_bf  = (bf16*) alloc((size_t)C * C * 2);
  bf16*  bigT_bf  = (bf16*) alloc((size_t)B * 512 * 4 * D * 2);      // [x_t|x1|x2|x3]
  bf16*  bigV_bf  = (bf16*) alloc((size_t)B * 512 * 4 * C * 2);      // [xp|x1|x2|x3]
  bf16*  xtime_bf = (bf16*) alloc((size_t)B * 512 * C * 2);
  bf16*  xvar_bf  = (bf16*) alloc((size_t)B * 512 * C * 2);
  float* xs_f32   = (float*)alloc((size_t)B * 512 * S * 4);          // pre-softmax maps

  // ---- output offsets (return order: time_list, var_list, time_consis, var_consis) ----
  const size_t tl_off[2] = {0ull, 16777216ull};
  const size_t vl_off[2] = {33554432ull, 50331648ull};
  const size_t tc_off[2] = {67108864ull, 67633152ull};
  const size_t vc_off[2] = {67895296ull, 68419584ull};

  auto gemm = [&](const bf16* A, long long lda, long long bsA,
                  const bf16* Bm, long long ldb, long long bsB,
                  float* Cf, long long ldcf, long long bsCf,
                  bf16* Cb, long long ldcb, long long bsCb,
                  int M, int N, int K, int batch,
                  const float* bias, int ep, int Lpe, int Dpe) {
    dim3 g((unsigned)(N / 64), (unsigned)(M / 64), (unsigned)batch);
    k_gemm_bf16<<<g, dim3(128), 0, stream>>>(A, lda, bsA, Bm, ldb, bsB,
                                             Cf, ldcf, bsCf, Cb, ldcb, bsCb,
                                             M, N, K, bias, ep, Lpe, Dpe);
  };

  // weights -> bf16 (once)
  k_f32_to_bf16<<<(3 * C * D + 255) / 256, 256, 0, stream>>>(conv_w, convw_bf, 3 * C * D);
  k_f32_to_bf16<<<(4 * D * C + 255) / 256, 256, 0, stream>>>(tg_w, tgw_bf, 4 * D * C);
  k_f32_to_bf16<<<(4 * C * C + 255) / 256, 256, 0, stream>>>(vg_w, vgw_bf, 4 * C * C);

  for (int i = 0; i < 2; ++i) {
    const int p = (i == 0) ? 2 : 4;
    const int L = S / p;
    const int BL = B * L;

    k_patch_mean<<<(BL * C + 255) / 256, 256, 0, stream>>>(x, xp_f32, bigV_bf, B, S, C, L, p);
    k_gather_conv_rows<<<(BL * 3 * C + 255) / 256, 256, 0, stream>>>(xp_f32, aexp_bf, B, L, C);
    k_f32_to_bf16<<<(C * S + 255) / 256, 256, 0, stream>>>(shv[i], sh_bf, C * S);
    k_f32_to_bf16<<<(L * L + 255) / 256, 256, 0, stream>>>(tvec[i], tv_bf, L * L);

    // token embed + positional embedding -> bigT slot 0 (bf16)
    gemm(aexp_bf, 3 * C, 0, convw_bf, D, 0,
         nullptr, 0, 0, bigT_bf, 4 * D, 0,
         BL, D, 3 * C, 1, nullptr, /*ep=*/2, L, D);

    // time adjacency: gelu(T@T) then row-softmax, stored transposed (A^T for diffusion)
    gemm(tv_bf, L, 0, tv_bf, L, 0,
         tg_f32, L, 0, nullptr, 0, 0,
         L, L, L, 1, nullptr, /*ep=*/1, 0, 0);
    k_softmax_transpose<<<L, 256, 0, stream>>>(tg_f32, tadjT_bf, L);

    // var adjacency (top-6 masked double-softmax)
    k_var_adj<<<1, 64, 0, stream>>>(vvec[i], vadj_bf);

    // time diffusion: slot_j = A^T . slot_{j-1}   (batched over B, shared A)
    for (int j = 1; j <= 3; ++j)
      gemm(tadjT_bf, L, 0,
           bigT_bf + (size_t)(j - 1) * D, 4 * D, (long long)L * 4 * D,
           nullptr, 0, 0,
           bigT_bf + (size_t)j * D, 4 * D, (long long)L * 4 * D,
           L, D, L, B, nullptr, 0, 0, 0);

    // var diffusion: slot_j = slot_{j-1} . Avar
    for (int j = 1; j <= 3; ++j)
      gemm(bigV_bf + (size_t)(j - 1) * C, 4 * C, (long long)L * 4 * C,
           vadj_bf, C, 0,
           nullptr, 0, 0,
           bigV_bf + (size_t)j * C, 4 * C, (long long)L * 4 * C,
           L, C, C, B, nullptr, 0, 0, 0);

    // final projections: gelu(concat @ W + b); f32 -> consistency outputs, bf16 -> next GEMM
    gemm(bigT_bf, 4 * D, 0, tgw_bf, C, 0,
         out + tc_off[i], C, 0, xtime_bf, C, 0,
         BL, C, 4 * D, 1, tg_b, /*ep=*/3, 0, 0);
    gemm(bigV_bf, 4 * C, 0, vgw_bf, C, 0,
         out + vc_off[i], C, 0, xvar_bf, C, 0,
         BL, C, 4 * C, 1, vg_b, /*ep=*/3, 0, 0);

    // attention maps: (x @ shared) -> tiled softmax into d_out
    gemm(xtime_bf, C, 0, sh_bf, S, 0,
         xs_f32, S, 0, nullptr, 0, 0,
         BL, S, C, 1, nullptr, 0, 0, 0);
    k_softmax_tile<<<BL, 256, 0, stream>>>(xs_f32, out + tl_off[i], L, S, p);

    gemm(xvar_bf, C, 0, sh_bf, S, 0,
         xs_f32, S, 0, nullptr, 0, 0,
         BL, S, C, 1, nullptr, 0, 0, 0);
    k_softmax_tile<<<BL, 256, 0, stream>>>(xs_f32, out + vl_off[i], L, S, p);
  }
}
